// Chord_Note_GRU_61177514165035
// MI455X (gfx1250) — matrix operand order repaired and checked by reading, not compiled, not run on hardware
//
#include <hip/hip_runtime.h>

// ---------------------------------------------------------------------------
// Chord/Note 5-layer GRU (B=32, T=512, D=H=512, C=150, V=832) for gfx1250.
// bf16 WMMA everywhere (v_wmma_f32_16x16x32_bf16), fp32 accumulate.
//   1) cast weights fp32->bf16 (fc_w zero-padded 150->160 rows)
//   2) embedding gather + ragged note mean -> x0 bf16 [16384,1024]
//   3) per layer: register-blocked WMMA GEMM xp = x @ w_ih^T + b_ih
//      (2 M-tiles x 4 N-tiles per wave -> 8 WMMA per 6 fragment loads)
//   4) per layer: persistent-grid recurrent scan; w_hh slice staged in LDS,
//      global barrier per timestep (monotonic atomic counter + s_sleep)
//   5) WMMA GEMM fc -> d_out fp32 [16384,150]
// ---------------------------------------------------------------------------

typedef __bf16 bf16_t;
typedef __attribute__((ext_vector_type(16))) __bf16 v16bf;
typedef __attribute__((ext_vector_type(8)))  __bf16 v8bf;
typedef __attribute__((ext_vector_type(8)))  float  v8f;

#define BB   32
#define TT   512
#define HH   512
#define G3H  1536
#define BT   16384   // B*T

// ---- WMMA fragment helpers (k contiguous per lane half in A and B, wave32) ----------

// A: 16x32 bf16, row M = lane&15; lanes 0-15 hold K 0..15 +kBase,
// lanes 16-31 hold K 16..31 +kBase. Two 16B loads per lane.
__device__ inline v16bf load_a_frag(const bf16_t* __restrict__ X, int ld,
                                    int mBase, int kBase, int lane) {
  const int m  = mBase + (lane & 15);
  const int kb = kBase + ((lane >> 4) << 4);
  const bf16_t* p = X + (size_t)m * ld + kb;
  v8bf lo = *(const v8bf*)p;
  v8bf hi = *(const v8bf*)(p + 8);
  v16bf a;
#pragma unroll
  for (int i = 0; i < 8; ++i) { a[i] = lo[i]; a[i + 8] = hi[i]; }
  return a;
}

// B: 32x16 bf16 taken from row-major W[N,K] (computing X*W^T): col N = lane&15,
// lanes 0-15 hold K 0..15, lanes 16-31 hold K 16..31. One 32B load per lane.
__device__ inline v16bf load_b_frag(const bf16_t* __restrict__ W, int ldk,
                                    int nBase, int kBase, int lane) {
  const int n  = nBase + (lane & 15);
  const int kb = kBase + ((lane >> 4) << 4);
  return *(const v16bf*)(W + (size_t)n * ldk + kb);
}

__device__ inline v8f wmma_bf16(v16bf a, v16bf b, v8f c) {
  return __builtin_amdgcn_wmma_f32_16x16x32_bf16(
      false, a, false, b, (short)0, c, false, false);
}

__device__ inline float sigmoidf_(float x) { return 1.0f / (1.0f + __expf(-x)); }

// ---- weight casts ----------------------------------------------------------

__global__ void cast_bf16_kernel(const float* __restrict__ s,
                                 bf16_t* __restrict__ d, size_t n) {
  size_t i = (size_t)blockIdx.x * blockDim.x + threadIdx.x;
  const size_t stride = (size_t)gridDim.x * blockDim.x;
  for (; i < n; i += stride) d[i] = (bf16_t)s[i];
}

__global__ void cast_pad_fc_kernel(const float* __restrict__ s,
                                   bf16_t* __restrict__ d) {
  // s: [150,512] -> d: [160,512] zero padded
  size_t i = (size_t)blockIdx.x * blockDim.x + threadIdx.x;
  if (i >= (size_t)160 * 512) return;
  const int n = (int)(i >> 9);
  d[i] = (n < 150) ? (bf16_t)s[i] : (bf16_t)0.0f;
}

// ---- embedding + ragged note mean -----------------------------------------

__global__ __launch_bounds__(256) void embed_kernel(
    const int* __restrict__ note, const int* __restrict__ chord,
    const float* __restrict__ ct, const float* __restrict__ nt,
    bf16_t* __restrict__ x0) {
  const int row = blockIdx.x;  // b*T + t
  __shared__ int ids[16];
  __shared__ int s_chord;
  if (threadIdx.x < 16) ids[threadIdx.x] = note[(size_t)row * 16 + threadIdx.x];
  if (threadIdx.x == 0) s_chord = chord[row];
  __syncthreads();
  int cnt = 0;
#pragma unroll
  for (int k = 0; k < 16; ++k) {
    if (ids[k] == 0) break;
    ++cnt;
  }
  const float inv = 1.0f / (float)(cnt > 0 ? cnt : 1);
  const int c = s_chord;
  for (int d = threadIdx.x; d < HH; d += 256) {
    const float ce = (c != 0) ? ct[(size_t)c * HH + d] : 0.0f;
    float acc = 0.0f;
    for (int k = 0; k < cnt; ++k) acc += nt[(size_t)ids[k] * HH + d];
    x0[(size_t)row * 1024 + d]      = (bf16_t)ce;
    x0[(size_t)row * 1024 + HH + d] = (bf16_t)(acc * inv);
  }
}

// ---- register-blocked WMMA GEMM: Y[M,nCols] = X[M,K] * W[N,K]^T + bias -----
// Each wave: 2 M-tiles x 4 N-tiles (32x64 output), 8 WMMA per k-step from
// 6 fragment loads.  Block = 8 waves covers 32 N-tiles; grid.x covers M/32.

__global__ __launch_bounds__(256) void gemm_bias_kernel(
    const bf16_t* __restrict__ X, const bf16_t* __restrict__ W,
    const float* __restrict__ bias, float* __restrict__ Y,
    int K, int Nld, int nCols, int nTilesMax) {
  const int lane   = threadIdx.x & 31;
  const int wave   = threadIdx.x >> 5;
  const int mBase  = blockIdx.x << 5;                 // 2 M-tiles
  const int nTile0 = (blockIdx.y * 8 + wave) << 2;    // 4 N-tiles
  if (nTile0 >= nTilesMax) return;

  int ntc[4];
#pragma unroll
  for (int j = 0; j < 4; ++j) {
    int nt = nTile0 + j;
    ntc[j] = (nt < nTilesMax) ? nt : (nTilesMax - 1);  // clamp loads (stores guarded)
  }

  v8f acc[2][4] = {};
  for (int kt = 0; kt < K; kt += 32) {
    v16bf a0 = load_a_frag(X, K, mBase, kt, lane);
    v16bf a1 = load_a_frag(X, K, mBase + 16, kt, lane);
#pragma unroll
    for (int j = 0; j < 4; ++j) {
      v16bf b = load_b_frag(W, K, ntc[j] << 4, kt, lane);
      acc[0][j] = wmma_bf16(a0, b, acc[0][j]);
      acc[1][j] = wmma_bf16(a1, b, acc[1][j]);
    }
  }

  const int rowSub = ((lane >> 4) << 3);
#pragma unroll
  for (int j = 0; j < 4; ++j) {
    const int nt = nTile0 + j;
    if (nt >= nTilesMax) break;
    const int col = (nt << 4) + (lane & 15);
    if (col >= nCols) continue;
    const float bv = bias[col];
#pragma unroll
    for (int mi = 0; mi < 2; ++mi) {
      const int rowBase = mBase + (mi << 4) + rowSub;
#pragma unroll
      for (int i = 0; i < 8; ++i)
        Y[(size_t)(rowBase + i) * Nld + col] = acc[mi][j][i] + bv;
    }
  }
}

// ---- persistent recurrent scan for one GRU layer ---------------------------
// 16 blocks x 4 waves = 64 wave-jobs: (mTile in {0,1}) x (jTile in 0..31).
// Each wave owns 16 h-columns, computes r/z/n tiles; its block stages the
// needed 96 KB of w_hh (2 jTiles x 3 gates x 16 rows x 512 K, bf16) in LDS.

__global__ __launch_bounds__(128) void gru_scan_kernel(
    const float* __restrict__ xp,    // [B,T,3H] fp32 (x @ w_ih^T + b_ih)
    const bf16_t* __restrict__ whh,  // [3H,H] bf16 row-major
    const float* __restrict__ bhh,   // [3H]
    float* __restrict__ hA, float* __restrict__ hB,      // fp32 h ping-pong
    bf16_t* __restrict__ hbA, bf16_t* __restrict__ hbB,  // bf16 h ping-pong
    bf16_t* __restrict__ ybf,        // [B*T,H] bf16 layer output
    unsigned* __restrict__ gctr) {   // global barrier counter (init 0)
  extern __shared__ char smem_raw[];
  bf16_t* lds = (bf16_t*)smem_raw;   // 6 regions * (16 rows * 512) bf16

  const int tid  = threadIdx.x;
  const int lane = tid & 31;
  const int wave = tid >> 5;
  const int bid  = blockIdx.x;

  // Stage w_hh slices: region r = jt_local*3 + gate, 16B chunks.
  for (int c = tid; c < 6144; c += 128) {
    const int region = c >> 10;
    const int rem    = c & 1023;
    const int rowi   = rem >> 6;
    const int kc     = rem & 63;
    const int jt_l   = region / 3;
    const int gate   = region % 3;
    const int nrow   = gate * HH + (bid * 2 + jt_l) * 16 + rowi;
    *(v8bf*)(lds + region * 8192 + rowi * HH + kc * 8) =
        *(const v8bf*)(whh + (size_t)nrow * HH + kc * 8);
  }
  __syncthreads();

  const int jt_local = wave >> 1;
  const int mTile    = wave & 1;
  const int jTile    = bid * 2 + jt_local;
  const int col      = (jTile << 4) + (lane & 15);
  const int rowBase  = (mTile << 4) + ((lane >> 4) << 3);
  const float br = bhh[col];
  const float bz = bhh[HH + col];
  const float bn = bhh[2 * HH + col];
  const bf16_t* ldsR = lds + (jt_local * 3 + 0) * 8192;
  const bf16_t* ldsZ = lds + (jt_local * 3 + 1) * 8192;
  const bf16_t* ldsN = lds + (jt_local * 3 + 2) * 8192;
  const int bOff = (lane & 15) * HH + ((lane >> 4) << 4);

  for (int t = 0; t < TT; ++t) {
    const bf16_t* hin  = (t & 1) ? hbB : hbA;
    const float*  hfp  = (t & 1) ? hB  : hA;
    float*        hout = (t & 1) ? hA  : hB;
    bf16_t*       hbo  = (t & 1) ? hbA : hbB;

    v8f cr = {}, cz = {}, cn = {};
#pragma unroll 4
    for (int kt = 0; kt < HH; kt += 32) {
      v16bf a   = load_a_frag(hin, HH, mTile << 4, kt, lane);
      v16bf brf = *(const v16bf*)(ldsR + bOff + kt);
      v16bf bzf = *(const v16bf*)(ldsZ + bOff + kt);
      v16bf bnf = *(const v16bf*)(ldsN + bOff + kt);
      cr = wmma_bf16(a, brf, cr);
      cz = wmma_bf16(a, bzf, cz);
      cn = wmma_bf16(a, bnf, cn);
    }
#pragma unroll
    for (int i = 0; i < 8; ++i) {
      const int b = rowBase + i;
      const float* xpr = xp + ((size_t)b * TT + t) * G3H;
      const float r  = sigmoidf_(xpr[col]          + cr[i] + br);
      const float z  = sigmoidf_(xpr[HH + col]     + cz[i] + bz);
      const float nn = tanhf(xpr[2 * HH + col] + r * (cn[i] + bn));
      const float hp = hfp[b * HH + col];
      const float hn = (1.0f - z) * nn + z * hp;
      hout[b * HH + col] = hn;
      const bf16_t hb = (bf16_t)hn;
      hbo[b * HH + col] = hb;
      ybf[((size_t)b * TT + t) * HH + col] = hb;
    }

    // device-wide barrier between timesteps (monotonic counter, no reset)
    __threadfence();
    __syncthreads();
    if (tid == 0) {
      const unsigned target = (unsigned)(t + 1) * gridDim.x;
      __hip_atomic_fetch_add(gctr, 1u, __ATOMIC_ACQ_REL, __HIP_MEMORY_SCOPE_AGENT);
      while (__hip_atomic_load(gctr, __ATOMIC_ACQUIRE, __HIP_MEMORY_SCOPE_AGENT) < target)
        __builtin_amdgcn_s_sleep(2);
    }
    __syncthreads();
  }
}

// ---------------------------------------------------------------------------

extern "C" void kernel_launch(void* const* d_in, const int* in_sizes, int n_in,
                              void* d_out, int out_size, void* d_ws, size_t ws_size,
                              hipStream_t stream) {
  const int*   note        = (const int*)d_in[0];
  const int*   chord       = (const int*)d_in[1];
  const float* chord_table = (const float*)d_in[2];
  const float* note_table  = (const float*)d_in[3];
  const float* w_ih0       = (const float*)d_in[4];
  const float* w_hh0       = (const float*)d_in[5];
  const float* b_ih0       = (const float*)d_in[6];
  const float* b_hh0       = (const float*)d_in[7];
  const float* w_ih_r      = (const float*)d_in[8];
  const float* w_hh_r      = (const float*)d_in[9];
  const float* b_ih_r      = (const float*)d_in[10];
  const float* b_hh_r      = (const float*)d_in[11];
  const float* fc_w        = (const float*)d_in[12];
  const float* fc_b        = (const float*)d_in[13];
  float* out = (float*)d_out;

  char* ws = (char*)d_ws;
  auto alloc = [&](size_t bytes) -> char* {
    char* p = ws;
    ws += (bytes + 255) & ~(size_t)255;
    return p;
  };
  bf16_t* wih0_bf = (bf16_t*)alloc((size_t)G3H * 1024 * 2);
  bf16_t* wihr_bf = (bf16_t*)alloc((size_t)4 * G3H * HH * 2);
  bf16_t* whh_bf  = (bf16_t*)alloc((size_t)5 * G3H * HH * 2);
  bf16_t* fcw_bf  = (bf16_t*)alloc((size_t)160 * HH * 2);
  bf16_t* x0_bf   = (bf16_t*)alloc((size_t)BT * 1024 * 2);
  bf16_t* ybf0    = (bf16_t*)alloc((size_t)BT * HH * 2);
  bf16_t* ybf1    = (bf16_t*)alloc((size_t)BT * HH * 2);
  float*  xp      = (float*)alloc((size_t)BT * G3H * 4);
  float*  hA      = (float*)alloc((size_t)BB * HH * 4);
  float*  hB      = (float*)alloc((size_t)BB * HH * 4);
  bf16_t* hbA     = (bf16_t*)alloc((size_t)BB * HH * 2);
  bf16_t* hbB     = (bf16_t*)alloc((size_t)BB * HH * 2);
  unsigned* gctr  = (unsigned*)alloc(256);

  // 1) weight casts to bf16
  cast_bf16_kernel<<<512, 256, 0, stream>>>(w_ih0, wih0_bf, (size_t)G3H * 1024);
  cast_bf16_kernel<<<512, 256, 0, stream>>>(w_ih_r, wihr_bf, (size_t)4 * G3H * HH);
  cast_bf16_kernel<<<512, 256, 0, stream>>>(w_hh0, whh_bf, (size_t)G3H * HH);
  cast_bf16_kernel<<<512, 256, 0, stream>>>(w_hh_r, whh_bf + (size_t)G3H * HH,
                                            (size_t)4 * G3H * HH);
  cast_pad_fc_kernel<<<(160 * HH + 255) / 256, 256, 0, stream>>>(fc_w, fcw_bf);

  // 2) embedding
  embed_kernel<<<BT, 256, 0, stream>>>(note, chord, chord_table, note_table, x0_bf);

  // 3) GRU layers
  for (int l = 0; l < 5; ++l) {
    bf16_t* yout = (l & 1) ? ybf1 : ybf0;
    const bf16_t* X   = (l == 0) ? x0_bf : ((l & 1) ? ybf0 : ybf1);
    const int K       = (l == 0) ? 1024 : HH;
    const bf16_t* Wih = (l == 0) ? wih0_bf : wihr_bf + (size_t)(l - 1) * G3H * HH;
    const float* bih  = (l == 0) ? b_ih0 : b_ih_r + (size_t)(l - 1) * G3H;
    const float* bhh  = (l == 0) ? b_hh0 : b_hh_r + (size_t)(l - 1) * G3H;
    const bf16_t* Whh = whh_bf + (size_t)l * G3H * HH;

    // 96 N-tiles / (8 waves * 4 tiles) = 3 blocks in y; 32 rows per block in x
    gemm_bias_kernel<<<dim3(BT / 32, 3), 256, 0, stream>>>(
        X, Wih, bih, xp, K, G3H, G3H, 96);

    hipMemsetAsync(hA, 0, (size_t)BB * HH * 4, stream);
    hipMemsetAsync(hbA, 0, (size_t)BB * HH * 2, stream);
    hipMemsetAsync(gctr, 0, sizeof(unsigned), stream);

    gru_scan_kernel<<<16, 128, 96 * 1024, stream>>>(
        xp, Whh, bhh, hA, hB, hbA, hbB, yout, gctr);
  }

  // 4) final FC: layer-4 output lives in ybf0 (10 N-tiles, stores guarded)
  gemm_bias_kernel<<<dim3(BT / 32, 1), 256, 0, stream>>>(
      ybf0, fcw_bf, fc_b, out, HH, 150, 150, 10);
}
